// SharedEdgeNet_60189671686198
// MI455X (gfx1250) — compile-verified, matrix operand order checked by reading, not run on hardware
//
#include <hip/hip_runtime.h>
#include <hip/hip_bf16.h>

typedef _Float16 v16h __attribute__((ext_vector_type(16)));
typedef float    v8f  __attribute__((ext_vector_type(8)));

#define INIT_MAP 0x007FFFFFu   // map(-inf)

__device__ __forceinline__ unsigned map_f(float f) {
    unsigned b = __float_as_uint(f);
    return (b & 0x80000000u) ? ~b : (b | 0x80000000u);
}
__device__ __forceinline__ float unmap_f(unsigned u) {
    unsigned b = (u & 0x80000000u) ? (u ^ 0x80000000u) : ~u;
    return __uint_as_float(b);
}
// decode node feature written by scatter-max; never-touched -> 0 (isfinite fixup)
__device__ __forceinline__ float h_decode(unsigned u) {
    return (u == INIT_MAP) ? 0.0f : unmap_f(u);
}
// K index inside a 16x32 f16 A/B fragment: half i (0..15), lane l (0..31)
__device__ __forceinline__ int frag_k(int i, int l) {
    return ((i < 8) ? i : i + 8) + 8 * (l >> 4);
}

// ---------------------------------------------------------------------------
// enc node transform: pq[i][0..63]=p=x@ (W1a-W1b)+b1 ; pq[i][64..127]=q=x@W1b
// F_IN = 4, scalar kernel (negligible FLOPs).
// ---------------------------------------------------------------------------
__global__ void enc_node_kernel(const float* __restrict__ x,
                                const float* __restrict__ W1,  // [8][64]
                                const float* __restrict__ b1,  // [64]
                                float* __restrict__ pq, int N) {
    __shared__ float sW[8 * 64];
    __shared__ float sB[64];
    int tid = threadIdx.x;
    for (int j = tid; j < 512; j += blockDim.x) sW[j] = W1[j];
    for (int j = tid; j < 64;  j += blockDim.x) sB[j] = b1[j];
    __syncthreads();
    int gid = blockIdx.x * blockDim.x + tid;
    int i = gid >> 6, n = gid & 63;
    if (i >= N) return;
    float x0 = x[i*4+0], x1 = x[i*4+1], x2 = x[i*4+2], x3 = x[i*4+3];
    float q = x0*sW[4*64+n] + x1*sW[5*64+n] + x2*sW[6*64+n] + x3*sW[7*64+n];
    float p = sB[n] + x0*sW[0*64+n] + x1*sW[1*64+n] + x2*sW[2*64+n] + x3*sW[3*64+n] - q;
    pq[i*128 + n]      = p;
    pq[i*128 + 64 + n] = q;
}

// ---------------------------------------------------------------------------
// core/dec node transform via WMMA: h(raw scatter buffer) -> pq[N][128]
// Bc[64][128]: cols 0..63 = W1a - W1b (p, +b1), cols 64..127 = W1b (q)
// 1 wave = 16 nodes; 8 waves/block; Bc staged in LDS as 16 pre-swizzled frags.
// ---------------------------------------------------------------------------
__global__ void node_wmma_kernel(const unsigned* __restrict__ hraw, // [N][64]
                                 const float* __restrict__ W1,      // [128][64]
                                 const float* __restrict__ b1,      // [64]
                                 float* __restrict__ pq, int N) {
    __shared__ v16h sB[16 * 32];   // frag f = kt*8+nt ; kt in {0,1}, nt in 0..7
    int tid = threadIdx.x;
    for (int j = tid; j < 16 * 32 * 16; j += blockDim.x) {
        int f = j >> 9, l = (j >> 4) & 31, i = j & 15;
        int kt = f >> 3, nt = f & 7;
        int n = nt * 16 + (l & 15);
        int k = kt * 32 + frag_k(i, l);
        float v = (n < 64) ? (W1[k*64 + n] - W1[(64+k)*64 + n])
                           : W1[(64+k)*64 + (n - 64)];
        ((_Float16*)&sB[f*32 + l])[i] = (_Float16)v;
    }
    __syncthreads();

    int wave = tid >> 5, lane = tid & 31;
    int tile = blockIdx.x * 8 + wave;
    int row = tile * 16 + (lane & 15);
    if (row >= N) row = N - 1;                 // idempotent duplicate work

    v16h a[2];
    #pragma unroll
    for (int kt = 0; kt < 2; ++kt) {
        int k1 = kt * 32 + 8 * (lane >> 4);
        const unsigned* hp = hraw + row * 64;
        uint4 u0 = *(const uint4*)(hp + k1);
        uint4 u1 = *(const uint4*)(hp + k1 + 4);
        uint4 u2 = *(const uint4*)(hp + k1 + 16);
        uint4 u3 = *(const uint4*)(hp + k1 + 20);
        a[kt][0]=(_Float16)h_decode(u0.x); a[kt][1]=(_Float16)h_decode(u0.y);
        a[kt][2]=(_Float16)h_decode(u0.z); a[kt][3]=(_Float16)h_decode(u0.w);
        a[kt][4]=(_Float16)h_decode(u1.x); a[kt][5]=(_Float16)h_decode(u1.y);
        a[kt][6]=(_Float16)h_decode(u1.z); a[kt][7]=(_Float16)h_decode(u1.w);
        a[kt][8]=(_Float16)h_decode(u2.x); a[kt][9]=(_Float16)h_decode(u2.y);
        a[kt][10]=(_Float16)h_decode(u2.z); a[kt][11]=(_Float16)h_decode(u2.w);
        a[kt][12]=(_Float16)h_decode(u3.x); a[kt][13]=(_Float16)h_decode(u3.y);
        a[kt][14]=(_Float16)h_decode(u3.z); a[kt][15]=(_Float16)h_decode(u3.w);
    }
    #pragma unroll
    for (int nt = 0; nt < 8; ++nt) {
        v8f acc = {};
        acc = __builtin_amdgcn_wmma_f32_16x16x32_f16(false, a[0], false, sB[(0*8+nt)*32 + lane], (short)0, acc, false, false);
        acc = __builtin_amdgcn_wmma_f32_16x16x32_f16(false, a[1], false, sB[(1*8+nt)*32 + lane], (short)0, acc, false, false);
        int n = nt * 16 + (lane & 15);
        float bias = (nt < 4) ? b1[n] : 0.0f;
        #pragma unroll
        for (int r = 0; r < 8; ++r) {
            int m = r + 8 * (lane >> 4);
            int nr = tile * 16 + m;
            if (nr >= N) nr = N - 1;           // same value -> benign race
            pq[nr * 128 + n] = acc[r] + bias;
        }
    }
}

// ---------------------------------------------------------------------------
// edge MLP (second GEMM) + scatter-max via WMMA:
// act[e] = relu(pq[dst][0:64] + pq[src][64:128]); m = act@W2 + b2
// atomicMax(u32-mapped) into hnext[dst][64]. 1 wave = 16 edges.
// ---------------------------------------------------------------------------
__global__ void edge_wmma_kernel(const float* __restrict__ pq,
                                 const int* __restrict__ idx,   // [2][E] (src,dst)
                                 const float* __restrict__ W2,  // [64][64]
                                 const float* __restrict__ b2,  // [64]
                                 unsigned* __restrict__ hnext,  // [N][64] mapped
                                 int E) {
    __shared__ v16h sB[8 * 32];    // frag f = kt*4+nt
    int tid = threadIdx.x;
    for (int j = tid; j < 8 * 32 * 16; j += blockDim.x) {
        int f = j >> 9, l = (j >> 4) & 31, i = j & 15;
        int kt = f >> 2, nt = f & 3;
        int n = nt * 16 + (l & 15);
        int k = kt * 32 + frag_k(i, l);
        ((_Float16*)&sB[f*32 + l])[i] = (_Float16)W2[k*64 + n];
    }
    __syncthreads();

    int wave = tid >> 5, lane = tid & 31;
    int tile = blockIdx.x * 8 + wave;
    int e = tile * 16 + (lane & 15);
    if (e >= E) e = E - 1;                     // max is idempotent
    int es = idx[e], ed = idx[E + e];
    const float* pd = pq + ed * 128;           // p(dst)
    const float* ps = pq + es * 128 + 64;      // q(src)

    v16h a[2];
    #pragma unroll
    for (int kt = 0; kt < 2; ++kt) {
        int k1 = kt * 32 + 8 * (lane >> 4);
        float4 d0 = *(const float4*)(pd + k1),      s0 = *(const float4*)(ps + k1);
        float4 d1 = *(const float4*)(pd + k1 + 4),  s1 = *(const float4*)(ps + k1 + 4);
        float4 d2 = *(const float4*)(pd + k1 + 16), s2 = *(const float4*)(ps + k1 + 16);
        float4 d3 = *(const float4*)(pd + k1 + 20), s3 = *(const float4*)(ps + k1 + 20);
        a[kt][0]=(_Float16)fmaxf(d0.x+s0.x,0.f); a[kt][1]=(_Float16)fmaxf(d0.y+s0.y,0.f);
        a[kt][2]=(_Float16)fmaxf(d0.z+s0.z,0.f); a[kt][3]=(_Float16)fmaxf(d0.w+s0.w,0.f);
        a[kt][4]=(_Float16)fmaxf(d1.x+s1.x,0.f); a[kt][5]=(_Float16)fmaxf(d1.y+s1.y,0.f);
        a[kt][6]=(_Float16)fmaxf(d1.z+s1.z,0.f); a[kt][7]=(_Float16)fmaxf(d1.w+s1.w,0.f);
        a[kt][8]=(_Float16)fmaxf(d2.x+s2.x,0.f); a[kt][9]=(_Float16)fmaxf(d2.y+s2.y,0.f);
        a[kt][10]=(_Float16)fmaxf(d2.z+s2.z,0.f); a[kt][11]=(_Float16)fmaxf(d2.w+s2.w,0.f);
        a[kt][12]=(_Float16)fmaxf(d3.x+s3.x,0.f); a[kt][13]=(_Float16)fmaxf(d3.y+s3.y,0.f);
        a[kt][14]=(_Float16)fmaxf(d3.z+s3.z,0.f); a[kt][15]=(_Float16)fmaxf(d3.w+s3.w,0.f);
    }

    int dv[8];                                  // dst per accumulator row
    #pragma unroll
    for (int r = 0; r < 8; ++r) {
        int er = tile * 16 + r + 8 * (lane >> 4);
        if (er >= E) er = E - 1;
        dv[r] = idx[E + er];
    }
    #pragma unroll
    for (int nt = 0; nt < 4; ++nt) {
        v8f acc = {};
        acc = __builtin_amdgcn_wmma_f32_16x16x32_f16(false, a[0], false, sB[(0*4+nt)*32 + lane], (short)0, acc, false, false);
        acc = __builtin_amdgcn_wmma_f32_16x16x32_f16(false, a[1], false, sB[(1*4+nt)*32 + lane], (short)0, acc, false, false);
        int n = nt * 16 + (lane & 15);
        float bias = b2[n];
        #pragma unroll
        for (int r = 0; r < 8; ++r)
            atomicMax(hnext + dv[r] * 64 + n, map_f(acc[r] + bias));
    }
}

// ---------------------------------------------------------------------------
// dec edge: scalar dot (H->1) + scatter-max into out_raw[N]
// ---------------------------------------------------------------------------
__global__ void dec_edge_kernel(const float* __restrict__ pq,
                                const int* __restrict__ idx,
                                const float* __restrict__ w2,  // [64]
                                const float* __restrict__ b2,  // [1]
                                unsigned* __restrict__ outraw, int E) {
    __shared__ float sw[64];
    __shared__ float sb;
    int tid = threadIdx.x;
    if (tid < 64) sw[tid] = w2[tid];
    if (tid == 0) sb = b2[0];
    __syncthreads();
    int e = blockIdx.x * blockDim.x + tid;
    if (e >= E) return;
    int es = idx[e], ed = idx[E + e];
    const float* pd = pq + ed * 128;
    const float* ps = pq + es * 128 + 64;
    float acc = sb;
    #pragma unroll
    for (int k = 0; k < 64; k += 4) {
        float4 d = *(const float4*)(pd + k);
        float4 s = *(const float4*)(ps + k);
        acc += fmaxf(d.x + s.x, 0.f) * sw[k + 0];
        acc += fmaxf(d.y + s.y, 0.f) * sw[k + 1];
        acc += fmaxf(d.z + s.z, 0.f) * sw[k + 2];
        acc += fmaxf(d.w + s.w, 0.f) * sw[k + 3];
    }
    atomicMax(outraw + ed, map_f(acc));
}

__global__ void fill_u32_kernel(unsigned* __restrict__ p, unsigned v, int n) {
    int i = blockIdx.x * blockDim.x + threadIdx.x;
    if (i < n) p[i] = v;
}

__global__ void finalize_kernel(const unsigned* __restrict__ outraw,
                                float* __restrict__ out, int N) {
    int i = blockIdx.x * blockDim.x + threadIdx.x;
    if (i < N) out[i] = h_decode(outraw[i]);
}

// ---------------------------------------------------------------------------
extern "C" void kernel_launch(void* const* d_in, const int* in_sizes, int n_in,
                              void* d_out, int out_size, void* d_ws, size_t ws_size,
                              hipStream_t stream) {
    const float* x     = (const float*)d_in[0];
    const int*   idx   = (const int*)  d_in[1];   // [2][E]
    const float* eattr = (const float*)d_in[2];
    const float* encW1 = (const float*)d_in[3];
    const float* encb1 = (const float*)d_in[4];
    const float* encW2 = (const float*)d_in[5];
    const float* encb2 = (const float*)d_in[6];
    const float* coreW1 = (const float*)d_in[7];
    const float* coreb1 = (const float*)d_in[8];
    const float* coreW2 = (const float*)d_in[9];
    const float* coreb2 = (const float*)d_in[10];
    const float* decW1 = (const float*)d_in[11];
    const float* decb1 = (const float*)d_in[12];
    const float* decW2 = (const float*)d_in[13];
    const float* decb2 = (const float*)d_in[14];

    const int N = in_sizes[0] / 4;
    const int E = in_sizes[1] / 2;

    char* w = (char*)d_ws;
    float*    pq     = (float*)w;                                    // N*128 f32
    unsigned* hraw   = (unsigned*)(w + (size_t)N * 128 * 4);         // N*64 u32
    unsigned* outraw = (unsigned*)(w + (size_t)N * 128 * 4 + (size_t)N * 64 * 4);

    const int nodeTiles = (N + 15) / 16;
    const int edgeTiles = (E + 15) / 16;
    const int nodeBlocks = (nodeTiles + 7) / 8;
    const int edgeBlocks = (edgeTiles + 7) / 8;
    const int fillBlocksH = (N * 64 + 255) / 256;

    // encoder layer
    enc_node_kernel<<<(N * 64 + 255) / 256, 256, 0, stream>>>(x, encW1, encb1, pq, N);
    fill_u32_kernel<<<fillBlocksH, 256, 0, stream>>>(hraw, INIT_MAP, N * 64);
    edge_wmma_kernel<<<edgeBlocks, 256, 0, stream>>>(pq, idx, encW2, encb2, hraw, E);

    // 8 core layers (shared weights)
    for (int it = 0; it < 8; ++it) {
        node_wmma_kernel<<<nodeBlocks, 256, 0, stream>>>(hraw, coreW1, coreb1, pq, N);
        fill_u32_kernel<<<fillBlocksH, 256, 0, stream>>>(hraw, INIT_MAP, N * 64);
        edge_wmma_kernel<<<edgeBlocks, 256, 0, stream>>>(pq, idx, coreW2, coreb2, hraw, E);
    }

    // decoder layer
    node_wmma_kernel<<<nodeBlocks, 256, 0, stream>>>(hraw, decW1, decb1, pq, N);
    fill_u32_kernel<<<(N + 255) / 256, 256, 0, stream>>>(outraw, INIT_MAP, N);
    dec_edge_kernel<<<(E + 255) / 256, 256, 0, stream>>>(pq, idx, decW2, decb2, outraw, E);
    finalize_kernel<<<(N + 255) / 256, 256, 0, stream>>>(outraw, (float*)d_out, N);

    // second tuple output: edge_attr passthrough
    hipMemcpyAsync((float*)d_out + N, eattr, (size_t)E * 4 * sizeof(float),
                   hipMemcpyDeviceToDevice, stream);
}
